// NestedParallelBlock_29454885716627
// MI455X (gfx1250) — matrix operand order repaired and checked by reading, not compiled, not run on hardware
//
#include <hip/hip_runtime.h>
#include <hip/hip_bf16.h>
#include <math.h>

// ---------------------------------------------------------------------------
// MI455X (gfx1250): wave32, WMMA bf16 (fp32 accum) for all matrix math,
// TDM (tensor_load_to_lds) for GEMM tile staging where the toolchain has it.
// Problem is compute-bound (~165 GFLOP vs ~100MB unique bytes -> ~4us HBM
// floor at 23.3TB/s), so everything dense runs on v_wmma_f32_16x16x32_bf16.
// ---------------------------------------------------------------------------

typedef __attribute__((ext_vector_type(16))) __bf16 v16bf;
typedef __attribute__((ext_vector_type(8)))  float  v8f;
typedef unsigned int uint32x4 __attribute__((ext_vector_type(4)));
typedef int          int32x4  __attribute__((ext_vector_type(4)));
typedef int          int32x8  __attribute__((ext_vector_type(8)));

#if __has_builtin(__builtin_amdgcn_tensor_load_to_lds) && \
    __has_builtin(__builtin_amdgcn_s_wait_tensorcnt)
#define HAVE_TDM 1
#else
#define HAVE_TDM 0
#endif

__device__ __forceinline__ v8f wmma_bf16(v16bf a, v16bf b, v8f c) {
  return __builtin_amdgcn_wmma_f32_16x16x32_bf16(false, a, false, b, (short)0, c,
                                                 false, false);
}

__device__ __forceinline__ void wavebar() {
#if defined(__HIP_DEVICE_COMPILE__)
  __builtin_amdgcn_wave_barrier();
#endif
}

#define DIM_   1024
#define HEADS  16
#define HD     64
#define SEQ    1024
#define BATCH  4
#define TOKENS 4096          // B * N
#define E_EXP  7168          // 3D + 4D
#define E_CON  2048          // 2D
#define K_CON  5120          // D + 4D
#define HROW   5120          // h = [attn_out | gelu(mlp)]

#if HAVE_TDM
// ---------------------------------------------------------------------------
// TDM: load a rows x 64(bf16) tile into LDS with an 8-element (16B) row pad,
// giving an LDS row stride of 72 bf16. D# per CDNA5 ISA ch.8:
//   group0: count=1 | lds_addr | global_addr(57b) | type=2
//   group1: data_size=2B, pad_enable, pad_interval=32 DWORDs (code 4),
//           pad_amount=4 DWORDs (code 3), dims/strides in elements
// Issued by one wave; completion via s_wait_tensorcnt + block barrier.
// ---------------------------------------------------------------------------
__device__ __forceinline__ void tdm_load_tile_128x64(const __bf16* gsrc,
                                                     unsigned ldsOff,
                                                     unsigned strideElems) {
  const unsigned long long ga = (unsigned long long)(size_t)gsrc;
  uint32x4 g0;
  g0[0] = 1u;                                          // count=1, user mode
  g0[1] = ldsOff;                                      // lds_addr (bytes)
  g0[2] = (unsigned)(ga & 0xffffffffu);                // global_addr[31:0]
  g0[3] = (unsigned)((ga >> 32) & 0x01ffffffu) | (2u << 30);  // [56:32]|type=2
  int32x8 g1;
  g1[0] = (1 << 16)      // data_size = 2 bytes
        | (1 << 20)      // pad_enable
        | (4 << 22)      // pad_interval: 32 DWORDs (= 64 bf16 row)
        | (3 << 25);     // pad_amount: 4 DWORDs (= 8 bf16 pad)
  g1[1] = (64u & 0xffffu) << 16;                       // tensor_dim0 lo16
  g1[2] = 0 | ((128u & 0xffffu) << 16);                // dim0 hi16 | dim1 lo16
  g1[3] = 0 | ((64u & 0xffffu) << 16);                 // dim1 hi16 | tile_dim0
  g1[4] = 128;                                         // tile_dim1 (tile_dim2=0)
  g1[5] = (int)strideElems;                            // tensor_dim0_stride lo32
  g1[6] = 0;
  g1[7] = 0;
  const int32x4 z4 = {0, 0, 0, 0};
#if __clang_major__ >= 23
  const int32x8 z8 = {0, 0, 0, 0, 0, 0, 0, 0};
  __builtin_amdgcn_tensor_load_to_lds(g0, g1, z4, z4, z8, 0);
#else
  __builtin_amdgcn_tensor_load_to_lds(g0, g1, z4, z4, 0);
#endif
}
#endif  // HAVE_TDM

// ---------------------------------------------------------------------------
// fp32 -> bf16 cast (weights): float4 in, packed 4xbf16 out
// ---------------------------------------------------------------------------
__global__ void cast_f32_to_bf16_kernel(const float4* __restrict__ in,
                                        uint2* __restrict__ out, size_t n4) {
  size_t i = (size_t)blockIdx.x * blockDim.x + threadIdx.x;
  const size_t stride = (size_t)gridDim.x * blockDim.x;
  for (; i < n4; i += stride) {
    const float4 v = in[i];
    __bf16 t[4] = {(__bf16)v.x, (__bf16)v.y, (__bf16)v.z, (__bf16)v.w};
    out[i] = *(const uint2*)t;
  }
}

// ---------------------------------------------------------------------------
// LayerNorm(x) * expert prefix mask -> bf16 ; one 256-thread block per token
// ---------------------------------------------------------------------------
__global__ __launch_bounds__(256) void ln_mask_cast_kernel(
    const float* __restrict__ x, const int* __restrict__ emask,
    const float* __restrict__ g, const float* __restrict__ b,
    __bf16* __restrict__ xn) {
  __shared__ float red[16];
  const int token = blockIdx.x;
  const int tid = threadIdx.x;
  const int lane = tid & 31, wv = tid >> 5;

  const float4 vx = ((const float4*)(x + (size_t)token * DIM_))[tid];
  float s  = vx.x + vx.y + vx.z + vx.w;
  float s2 = vx.x * vx.x + vx.y * vx.y + vx.z * vx.z + vx.w * vx.w;
#pragma unroll
  for (int off = 1; off < 32; off <<= 1) {
    s  += __shfl_xor(s,  off, 32);
    s2 += __shfl_xor(s2, off, 32);
  }
  if (lane == 0) { red[wv] = s; red[8 + wv] = s2; }
  __syncthreads();
  float ts = 0.f, ts2 = 0.f;
#pragma unroll
  for (int i = 0; i < 8; ++i) { ts += red[i]; ts2 += red[8 + i]; }

  const float mean = ts * (1.0f / DIM_);
  const float var  = ts2 * (1.0f / DIM_) - mean * mean;
  const float rstd = rsqrtf(var + 1e-5f);
  const int   de   = DIM_ >> (3 - emask[token]);

  const float vals[4] = {vx.x, vx.y, vx.z, vx.w};
#pragma unroll
  for (int j = 0; j < 4; ++j) {
    const int col = tid * 4 + j;
    float o = (vals[j] - mean) * rstd * g[col] + b[col];
    if (col >= de) o = 0.f;
    xn[(size_t)token * DIM_ + col] = (__bf16)o;
  }
}

// ---------------------------------------------------------------------------
// Tiled bf16 WMMA GEMM:  out[m,e] = sum_k A[m,k] * W[e,k]  (+ bias if e<lim)
// Block tile 128(M) x 128(E), K-step 64 (16 WMMAs per barrier pair).
// Tiles staged to LDS by the TDM when available (one wave issues two 128x64
// tensor_load_to_lds with HW row padding), else by vectorized b128 loads.
// ---------------------------------------------------------------------------
template <bool OUT_BF16>
__global__ __launch_bounds__(256) void gemm_bf16_kernel(
    const __bf16* __restrict__ A, const __bf16* __restrict__ W,
    const float* __restrict__ bias, int biasLimit, int K, int E,
    void* __restrict__ outP) {
  constexpr int BK = 64;
  constexpr int LDSTR = 72;    // 64 + 8 bf16 pad (matches TDM pad config)
  __shared__ __align__(16) __bf16 lA[128 * LDSTR];
  __shared__ __align__(16) __bf16 lB[128 * LDSTR];

  const int tid    = threadIdx.x;
  const int lane   = tid & 31;
  const int laneLo = lane & 15;
  const int kHalf  = (lane >> 4) << 4;
  const int hiRow  = (lane >> 4) << 3;
  const int wv     = tid >> 5;
  const int mO     = (wv & 3) * 32;
  const int eO     = (wv >> 2) * 64;
  const int eBase  = blockIdx.x * 128;
  const int mBase  = blockIdx.y * 128;

#if !HAVE_TDM
  const int srow  = tid >> 1;
  const int shalf = (tid & 1) * 32;
  const __bf16* gA = A + (size_t)(mBase + srow) * K + shalf;
  const __bf16* gW = W + (size_t)(eBase + srow) * K + shalf;
  __bf16* sA = &lA[srow * LDSTR + shalf];
  __bf16* sW = &lB[srow * LDSTR + shalf];
#endif

  v8f acc[2][4] = {};

  for (int k0 = 0; k0 < K; k0 += BK) {
#if HAVE_TDM
    if (wv == 0) {
      tdm_load_tile_128x64(A + (size_t)mBase * K + k0,
                           (unsigned)(size_t)(void*)lA, (unsigned)K);
      tdm_load_tile_128x64(W + (size_t)eBase * K + k0,
                           (unsigned)(size_t)(void*)lB, (unsigned)K);
      __builtin_amdgcn_s_wait_tensorcnt(0);
    }
    __syncthreads();
#else
#pragma unroll
    for (int i = 0; i < 4; ++i)
      *(uint4*)(sA + i * 8) = *(const uint4*)(gA + i * 8);
#pragma unroll
    for (int i = 0; i < 4; ++i)
      *(uint4*)(sW + i * 8) = *(const uint4*)(gW + i * 8);
    if (k0 + BK < K) {
      __builtin_prefetch(gA + BK, 0, 1);
      __builtin_prefetch(gW + BK, 0, 1);
    }
    __syncthreads();
#endif

    v16bf af[2][2], wf[2][4];
#pragma unroll
    for (int s = 0; s < 2; ++s) {
#pragma unroll
      for (int mi = 0; mi < 2; ++mi)
        af[s][mi] =
            *(const v16bf*)&lA[(mO + mi * 16 + laneLo) * LDSTR + s * 32 + kHalf];
#pragma unroll
      for (int ni = 0; ni < 4; ++ni)
        wf[s][ni] =
            *(const v16bf*)&lB[(eO + ni * 16 + laneLo) * LDSTR + s * 32 + kHalf];
    }
#pragma unroll
    for (int mi = 0; mi < 2; ++mi)
#pragma unroll
      for (int ni = 0; ni < 4; ++ni) {
        acc[mi][ni] = wmma_bf16(af[0][mi], wf[0][ni], acc[mi][ni]);
        acc[mi][ni] = wmma_bf16(af[1][mi], wf[1][ni], acc[mi][ni]);
      }
    __syncthreads();
#if !HAVE_TDM
    gA += BK;
    gW += BK;
#endif
  }

  const int colBase = eBase + eO + laneLo;
#pragma unroll
  for (int mi = 0; mi < 2; ++mi) {
#pragma unroll
    for (int ni = 0; ni < 4; ++ni) {
      const int col  = colBase + ni * 16;
      const float bv = (col < biasLimit) ? bias[col] : 0.f;
#pragma unroll
      for (int r = 0; r < 8; ++r) {
        const int rw = mBase + mO + mi * 16 + r + hiRow;
        const float val = acc[mi][ni][r] + bv;
        if constexpr (OUT_BF16)
          ((__bf16*)outP)[(size_t)rw * E + col] = (__bf16)val;
        else
          ((float*)outP)[(size_t)rw * E + col] = val;
      }
    }
  }
}

// ---------------------------------------------------------------------------
// Split y -> q (head-major), k = LN2 (head-major), v = LN2 (transposed per head),
// mlp -> exact GELU. Reproduces the reference input_bias quirk (mlp_bias was
// already added to y's first 4096 cols in the GEMM epilogue; the mlp slice
// adds mlp_bias again).
// ---------------------------------------------------------------------------
__global__ __launch_bounds__(256) void split_qkv_kernel(
    const __bf16* __restrict__ y, const float* __restrict__ mlpb,
    const float* __restrict__ g2, const float* __restrict__ b2,
    __bf16* __restrict__ qh, __bf16* __restrict__ kT,
    __bf16* __restrict__ vh, __bf16* __restrict__ hb) {
  __shared__ float red[16];
  const int token = blockIdx.x;
  const int bI = token >> 10, n = token & 1023;
  const int tid = threadIdx.x;
  const int lane = tid & 31, wv = tid >> 5;
  const __bf16* yr = y + (size_t)token * E_EXP;

#pragma unroll
  for (int j = 0; j < 4; ++j) {
    const int c = tid * 4 + j;
    const int h = c >> 6, hd = c & 63;
    qh[(((size_t)bI * HEADS + h) * SEQ + n) * HD + hd] = yr[c];
  }

  float kv[4], vv[4];
  float ks = 0.f, ks2 = 0.f, vs = 0.f, vs2 = 0.f;
#pragma unroll
  for (int j = 0; j < 4; ++j) {
    const int c = tid * 4 + j;
    kv[j] = (float)yr[DIM_ + c];
    vv[j] = (float)yr[2 * DIM_ + c];
    ks += kv[j]; ks2 += kv[j] * kv[j];
    vs += vv[j]; vs2 += vv[j] * vv[j];
  }
#pragma unroll
  for (int off = 1; off < 32; off <<= 1) {
    ks += __shfl_xor(ks, off, 32);
    ks2 += __shfl_xor(ks2, off, 32);
  }
  if (lane == 0) { red[wv] = ks; red[8 + wv] = ks2; }
  __syncthreads();
  float tks = 0.f, tks2 = 0.f;
#pragma unroll
  for (int i = 0; i < 8; ++i) { tks += red[i]; tks2 += red[8 + i]; }
  __syncthreads();
#pragma unroll
  for (int off = 1; off < 32; off <<= 1) {
    vs += __shfl_xor(vs, off, 32);
    vs2 += __shfl_xor(vs2, off, 32);
  }
  if (lane == 0) { red[wv] = vs; red[8 + wv] = vs2; }
  __syncthreads();
  float tvs = 0.f, tvs2 = 0.f;
#pragma unroll
  for (int i = 0; i < 8; ++i) { tvs += red[i]; tvs2 += red[8 + i]; }

  const float km = tks * (1.f / DIM_);
  const float kr = rsqrtf(tks2 * (1.f / DIM_) - km * km + 1e-5f);
  const float vm = tvs * (1.f / DIM_);
  const float vr = rsqrtf(tvs2 * (1.f / DIM_) - vm * vm + 1e-5f);

#pragma unroll
  for (int j = 0; j < 4; ++j) {
    const int c = tid * 4 + j;
    const int h = c >> 6, hd = c & 63;
    const float kn = (kv[j] - km) * kr * g2[c] + b2[c];
    const float vn = (vv[j] - vm) * vr * g2[c] + b2[c];
    kT[(((size_t)bI * HEADS + h) * SEQ + n) * HD + hd] = (__bf16)kn;
    vh[(((size_t)bI * HEADS + h) * HD + hd) * SEQ + n] = (__bf16)vn;
  }

#pragma unroll
  for (int j = 0; j < 16; ++j) {
    const int c = tid + 256 * j;
    const float xm = (float)yr[3 * DIM_ + c] + mlpb[c];
    const float ge = 0.5f * xm * (1.0f + erff(xm * 0.70710678118654752f));
    hb[(size_t)token * HROW + DIM_ + c] = (__bf16)ge;
  }
}

// ---------------------------------------------------------------------------
// Flash attention: 4 waves/block, each wave owns 16 query rows; keys streamed
// in chunks of 32. Row-sum of exp(S) computed by a P x ones WMMA (the result
// lands row-replicated in exactly the C-layout my l[] state uses), so only
// the row-max needs shuffle trees. P staged through per-wave LDS with a
// wave_barrier (no block barriers in this kernel at all).
// ---------------------------------------------------------------------------
__global__ __launch_bounds__(128) void attn_kernel(
    const __bf16* __restrict__ q, const __bf16* __restrict__ kT,
    const __bf16* __restrict__ v, __bf16* __restrict__ hb) {
  __shared__ __align__(16) __bf16 pScr[4][16][40];
  const int tid    = threadIdx.x;
  const int wv     = tid >> 5;
  const int lane   = tid & 31;
  const int laneLo = lane & 15;
  const int kHalf  = (lane >> 4) << 4;
  const int hiRow  = (lane >> 4) << 3;

  const int bh = blockIdx.z * HEADS + blockIdx.y;
  const int q0 = blockIdx.x * 64 + wv * 16;

  const __bf16* qB = q + (size_t)bh * SEQ * HD;
  const v16bf aQ0 = *(const v16bf*)(qB + (size_t)(q0 + laneLo) * HD + kHalf);
  const v16bf aQ1 = *(const v16bf*)(qB + (size_t)(q0 + laneLo) * HD + 32 + kHalf);

  const __bf16* kB = kT + (size_t)bh * HD * SEQ;
  const __bf16* vB = v + (size_t)bh * SEQ * HD;

  v16bf vones;
#pragma unroll
  for (int i = 0; i < 16; ++i) vones[i] = (__bf16)1.0f;

  v8f o0 = {}, o1 = {}, o2 = {}, o3 = {};
  float m[8], l[8];
#pragma unroll
  for (int r = 0; r < 8; ++r) { m[r] = -3.0e38f; l[r] = 0.f; }

  for (int c = 0; c < SEQ / 32; ++c) {
    const int key0 = c * 32;
    v8f s0 = {}, s1 = {};
    const v16bf bK0a = *(const v16bf*)(kB + (size_t)(key0 + laneLo) * HD + kHalf);
    const v16bf bK1a = *(const v16bf*)(kB + (size_t)(key0 + laneLo) * HD + 32 + kHalf);
    const v16bf bK0b = *(const v16bf*)(kB + (size_t)(key0 + 16 + laneLo) * HD + kHalf);
    const v16bf bK1b = *(const v16bf*)(kB + (size_t)(key0 + 16 + laneLo) * HD + 32 + kHalf);
    s0 = wmma_bf16(aQ0, bK0a, s0);
    s0 = wmma_bf16(aQ1, bK1a, s0);
    s1 = wmma_bf16(aQ0, bK0b, s1);
    s1 = wmma_bf16(aQ1, bK1b, s1);

#pragma unroll
    for (int r = 0; r < 8; ++r) {
      const float a = s0[r] * 0.125f;   // HEAD_DIM^-0.5
      const float b = s1[r] * 0.125f;
      float t = fmaxf(a, b);
#pragma unroll
      for (int off = 1; off < 16; off <<= 1) t = fmaxf(t, __shfl_xor(t, off, 32));
      const float nm = fmaxf(m[r], t);
      const float p0 = __expf(a - nm);
      const float p1 = __expf(b - nm);
      const float sc = __expf(m[r] - nm);
      l[r] *= sc;
      m[r] = nm;
      o0[r] *= sc; o1[r] *= sc; o2[r] *= sc; o3[r] *= sc;
      pScr[wv][r + hiRow][laneLo]      = (__bf16)p0;
      pScr[wv][r + hiRow][laneLo + 16] = (__bf16)p1;
    }
    wavebar();                       // intra-wave LDS publish (no instruction)
    const v16bf aP = *(const v16bf*)&pScr[wv][laneLo][kHalf];
    wavebar();

    v8f ls = {};
    ls = wmma_bf16(aP, vones, ls);   // row-sum of P, row-replicated in C layout
#pragma unroll
    for (int r = 0; r < 8; ++r) l[r] += ls[r];

    const v16bf bV0 = *(const v16bf*)(vB + (size_t)(0 + laneLo) * SEQ + key0 + kHalf);
    const v16bf bV1 = *(const v16bf*)(vB + (size_t)(16 + laneLo) * SEQ + key0 + kHalf);
    const v16bf bV2 = *(const v16bf*)(vB + (size_t)(32 + laneLo) * SEQ + key0 + kHalf);
    const v16bf bV3 = *(const v16bf*)(vB + (size_t)(48 + laneLo) * SEQ + key0 + kHalf);
    o0 = wmma_bf16(aP, bV0, o0);
    o1 = wmma_bf16(aP, bV1, o1);
    o2 = wmma_bf16(aP, bV2, o2);
    o3 = wmma_bf16(aP, bV3, o3);
  }

  const int hcolBase = blockIdx.y * HD;
#pragma unroll
  for (int r = 0; r < 8; ++r) {
    const float inv = 1.0f / l[r];
    const int tok = blockIdx.z * SEQ + q0 + r + hiRow;
    __bf16* dst = hb + (size_t)tok * HROW + hcolBase + laneLo;
    dst[0]  = (__bf16)(o0[r] * inv);
    dst[16] = (__bf16)(o1[r] * inv);
    dst[32] = (__bf16)(o2[r] * inv);
    dst[48] = (__bf16)(o3[r] * inv);
  }
}

// ---------------------------------------------------------------------------
// Final combine: expert mask on both halves, residual, (alpha*p + 1) gate.
// ---------------------------------------------------------------------------
__global__ __launch_bounds__(256) void combine_kernel(
    const float* __restrict__ ob, const float* __restrict__ x,
    const int* __restrict__ emask, const float* __restrict__ probs,
    const float* __restrict__ alpha, float* __restrict__ out) {
  const int token = blockIdx.x;
  const int tid = threadIdx.x;
  const int de = DIM_ >> (3 - emask[token]);
  const float gate = alpha[0] * probs[token] + 1.0f;
  const float* o = ob + (size_t)token * E_CON;
  const float* xr = x + (size_t)token * DIM_;
  float* dr = out + (size_t)token * DIM_;
#pragma unroll
  for (int j = 0; j < 4; ++j) {
    const int c = tid * 4 + j;
    const float fm = (c < de) ? 1.0f : 0.0f;
    dr[c] = o[c] * fm + xr[c] + gate * o[DIM_ + c] * fm;
  }
}

// ---------------------------------------------------------------------------
extern "C" void kernel_launch(void* const* d_in, const int* in_sizes, int n_in,
                              void* d_out, int out_size, void* d_ws,
                              size_t ws_size, hipStream_t stream) {
  (void)in_sizes; (void)n_in; (void)out_size; (void)ws_size;

  const float* x        = (const float*)d_in[0];
  const int*   emask    = (const int*)d_in[1];
  const float* probs    = (const float*)d_in[2];
  const float* wExp32   = (const float*)d_in[3];
  const float* mlp_bias = (const float*)d_in[4];
  const float* wCon32   = (const float*)d_in[5];
  const float* con_bias = (const float*)d_in[6];
  const float* g1       = (const float*)d_in[7];
  const float* b1       = (const float*)d_in[8];
  const float* g2       = (const float*)d_in[9];
  const float* b2       = (const float*)d_in[10];
  const float* alpha    = (const float*)d_in[11];

  char* ws = (char*)d_ws;
  size_t off = 0;
  auto alloc = [&](size_t bytes) -> void* {
    off = (off + 255) & ~(size_t)255;
    void* p = ws + off;
    off += bytes;
    return p;
  };

  __bf16* wExp = (__bf16*)alloc((size_t)E_EXP * DIM_ * 2);
  __bf16* wCon = (__bf16*)alloc((size_t)E_CON * K_CON * 2);
  __bf16* xn   = (__bf16*)alloc((size_t)TOKENS * DIM_ * 2);
  __bf16* y    = (__bf16*)alloc((size_t)TOKENS * E_EXP * 2);
  __bf16* qh   = (__bf16*)alloc((size_t)TOKENS * DIM_ * 2);
  __bf16* kT   = (__bf16*)alloc((size_t)TOKENS * DIM_ * 2);
  __bf16* vh   = (__bf16*)alloc((size_t)TOKENS * DIM_ * 2);
  __bf16* hb   = (__bf16*)alloc((size_t)TOKENS * HROW * 2);
  float*  ob   = (float*)alloc((size_t)TOKENS * E_CON * 4);

  cast_f32_to_bf16_kernel<<<1024, 256, 0, stream>>>(
      (const float4*)wExp32, (uint2*)wExp, (size_t)E_EXP * DIM_ / 4);
  cast_f32_to_bf16_kernel<<<1024, 256, 0, stream>>>(
      (const float4*)wCon32, (uint2*)wCon, (size_t)E_CON * K_CON / 4);

  ln_mask_cast_kernel<<<TOKENS, 256, 0, stream>>>(x, emask, g1, b1, xn);

  gemm_bf16_kernel<true><<<dim3(E_EXP / 128, TOKENS / 128), 256, 0, stream>>>(
      xn, wExp, mlp_bias, 4096, DIM_, E_EXP, (void*)y);

  split_qkv_kernel<<<TOKENS, 256, 0, stream>>>(y, mlp_bias, g2, b2, qh, kT, vh,
                                               hb);

  attn_kernel<<<dim3(SEQ / 64, HEADS, BATCH), 128, 0, stream>>>(qh, kT, vh, hb);

  gemm_bf16_kernel<false><<<dim3(E_CON / 128, TOKENS / 128), 256, 0, stream>>>(
      hb, wCon, con_bias, E_CON, K_CON, E_CON, (void*)ob);

  combine_kernel<<<TOKENS, 256, 0, stream>>>(ob, x, emask, probs, alpha,
                                             (float*)d_out);
}
